// ODE_65575560675715
// MI455X (gfx1250) — matrix operand order repaired and checked by reading, not compiled, not run on hardware
//
#include <hip/hip_runtime.h>
#include <hip/hip_bf16.h>

// Fused grouped MLP-ODE cell, per (n,g):
//   h1 = swish_b1( w1[g] (512x64)  @ x (64x64) )
//   h2 = swish_b2( w2[g] (512x512) @ h1 )
//   o  = tanh    ( w3[g] (64x512)  @ h2 )
// Fast path: weights pre-converted fp32->bf16 into d_ws in WMMA-fragment-
// swizzled layout; GEMMs use v_wmma_f32_16x16x32_bf16 with k-outer loops and
// 4x4 accumulator blocks (B fragments loaded once per k-step, not 4x).

typedef __bf16 bf16_t;
typedef __attribute__((ext_vector_type(16))) __bf16       v16bf;
typedef __attribute__((ext_vector_type(8)))  float        v8f;
typedef __attribute__((ext_vector_type(4)))  unsigned int v4u;

#define G_ 128
#define D_ 64
#define H_ 512
#define L_ 64

__device__ __forceinline__ unsigned int pack_bf16x2(float lo, float hi) {
  union { bf16_t h[2]; unsigned int u; } p;
  p.h[0] = (bf16_t)lo;
  p.h[1] = (bf16_t)hi;
  return p.u;
}

__device__ __forceinline__ float swishf(float x, float b) {
  return x / (1.0f + __expf(-b * x));
}

// Saturating-safe fast tanh: exact limits at +/-inf, two VALU + one v_exp.
__device__ __forceinline__ float tanh_fast(float x) {
  return 1.0f - 2.0f / (__expf(2.0f * x) + 1.0f);
}

__device__ __forceinline__ void wmma_bf16(v16bf a, v16bf b, v8f& acc) {
  acc = __builtin_amdgcn_wmma_f32_16x16x32_bf16(false, a, false, b, (short)0, acc,
                                                false, false);
}

// ---------------------------------------------------------------------------
// Fragment-swizzled storage: a 16x32 operand tile occupies 1KB; lane L's 16
// bf16 elements (32B) live at tile_base + L*32. Loader is mapping-agnostic.
// ---------------------------------------------------------------------------
template <typename PTR>
__device__ __forceinline__ v16bf load_frag_sw(PTR base, int tile, int lane) {
  const v4u* p = reinterpret_cast<const v4u*>(base + ((size_t)tile * 32 + lane) * 16);
  union { v16bf v; v4u q[2]; } f;
  f.q[0] = p[0];
  f.q[1] = p[1];
  return f.v;
}

// A-fragment loaders (overloaded on weight element type).
// bf16: pre-swizzled, two global_load_b128. float: fp32 row-major + in-reg cvt.
__device__ __forceinline__ v16bf load_a_any(const bf16_t* A, int ktiles, int m0,
                                            int ks, int lane) {
  return load_frag_sw(A, (m0 >> 4) * ktiles + ks, lane);
}

__device__ __forceinline__ v16bf load_a_any(const float* __restrict__ A, int ktiles,
                                            int m0, int ks, int lane) {
  const int lda  = ktiles * 32;
  const int half = lane >> 4;
  const int row  = m0 + (lane & 15);
  const float* base = A + (size_t)row * lda + (ks * 32 + half * 16);
  union { v16bf v; unsigned int w[8]; } f;
#pragma unroll
  for (int j = 0; j < 4; ++j) {
    float2 p0 = *reinterpret_cast<const float2*>(base + 2 * j);
    float2 p1 = *reinterpret_cast<const float2*>(base + 8 + 2 * j);
    f.w[j]     = pack_bf16x2(p0.x, p0.y);
    f.w[4 + j] = pack_bf16x2(p1.x, p1.y);
  }
  return f.v;
}

// Max-locality prefetch -> WGP scope ("pull into all cache levels on miss").
__device__ __forceinline__ void prefetch_a(const bf16_t* A, int ktiles, int m0,
                                           int ks, int lane) {
  __builtin_prefetch(A + ((size_t)((m0 >> 4) * ktiles + ks + 1) * 32 + lane) * 16, 0, 3);
}
__device__ __forceinline__ void prefetch_a(const float*, int, int, int, int) {}

// Swish + convert + store C tile into the NEXT gemm's B-swizzled LDS layout.
// B mapping: element(col,K): tile=(K>>5)*4 + (col>>4); lane=(col&15)+16*((K>>4)&1);
// in-lane bf16 offset = K&15. The 8 consecutive M of a C-register column are
// contiguous inside one target lane chunk -> one 16B ds_store per lane.
__device__ __forceinline__ void store_swish_sw(bf16_t* Bsw, int m0, int n0, v8f acc,
                                               float beta, int lane) {
  const int mbase = m0 + (lane >> 4) * 8;  // next-gemm K base
  const int tile  = (mbase >> 5) * 4 + (n0 >> 4);
  const int lb    = (lane & 15) + (((mbase >> 4) & 1) << 4);
  v4u packed;
#pragma unroll
  for (int r = 0; r < 4; ++r) {
    float v0 = swishf(acc[2 * r], beta);
    float v1 = swishf(acc[2 * r + 1], beta);
    packed[r] = pack_bf16x2(v0, v1);
  }
  *reinterpret_cast<v4u*>(Bsw + ((size_t)tile * 32 + lb) * 16 + (mbase & 15)) = packed;
}

// 512-row GEMM block: wave owns 64 M-rows (4 m-tiles) x all 64 cols (4 n-tiles).
// k-outer: B fragments loaded ONCE per k-step, shared across 4 m-tiles.
// 16 x v8f accumulators (128 VGPRs) -- occupancy is LDS-bound anyway.
template <typename AW>
__device__ __forceinline__ void gemm512_swish(const AW* Ag, int ktiles,
                                              const bf16_t* Bsw, bf16_t* Csw,
                                              float beta, int wv, int lane) {
  const int mrow0 = wv * 64;
  v8f acc[16] = {};
  for (int ks = 0; ks < ktiles; ++ks) {
    v16bf bfrag[4];
#pragma unroll
    for (int nt = 0; nt < 4; ++nt) bfrag[nt] = load_frag_sw(Bsw, ks * 4 + nt, lane);
#pragma unroll
    for (int mt = 0; mt < 4; ++mt) {
      const int m0 = mrow0 + mt * 16;
      v16bf a = load_a_any(Ag, ktiles, m0, ks, lane);
      if (ks + 1 < ktiles) prefetch_a(Ag, ktiles, m0, ks, lane);
#pragma unroll
      for (int nt = 0; nt < 4; ++nt) wmma_bf16(a, bfrag[nt], acc[mt * 4 + nt]);
    }
  }
#pragma unroll
  for (int mt = 0; mt < 4; ++mt)
#pragma unroll
    for (int nt = 0; nt < 4; ++nt)
      store_swish_sw(Csw, mrow0 + mt * 16, nt * 16, acc[mt * 4 + nt], beta, lane);
}

// ---------------------------------------------------------------------------
// Weight conversion: fp32 [G][M][K] row-major -> bf16 A-fragment-swizzled.
// A mapping (16-bit A 16x32): lane = r + 16*h (r=row&15), dword j (0..7) holds
// K = h*16+2j(+1). Tiles mt-major, kt-minor.
// ---------------------------------------------------------------------------
__global__ __launch_bounds__(256) void convert_swizzle_kernel(
    const float* __restrict__ src, bf16_t* __restrict__ dst, int K, int kshift) {
  const int gidx = blockIdx.y;
  const size_t per_group_dw = (size_t)gridDim.x * 256;  // M*K/2
  const size_t dd = (size_t)blockIdx.x * 256 + threadIdx.x;

  const int j    = (int)(dd & 7);
  const int lane = (int)((dd >> 3) & 31);
  const size_t tile = dd >> 8;
  const int kt = (int)(tile & ((1u << kshift) - 1));
  const int mt = (int)(tile >> kshift);
  const int r = lane & 15;
  const int h = lane >> 4;
  const int kk = h * 16 + 2 * j;

  const float* s = src + ((size_t)gidx * ((size_t)gridDim.x * 512 / K) + mt * 16 + r) * K +
                   kt * 32 + kk;
  const unsigned int p = pack_bf16x2(s[0], s[1]);
  reinterpret_cast<unsigned int*>(dst)[gidx * per_group_dw + dd] = p;
}

// ---------------------------------------------------------------------------
// Main kernel (templated on weight element type: bf16 = fast, float = fallback)
// ---------------------------------------------------------------------------
template <typename AW>
__global__ __launch_bounds__(256) void ode_fused_kernel_t(
    const float* __restrict__ z, const AW* __restrict__ w1,
    const AW* __restrict__ w2, const AW* __restrict__ w3,
    const float* __restrict__ beta1, const float* __restrict__ beta2,
    float* __restrict__ out) {
  extern __shared__ char smem_raw[];
  bf16_t* xsw  = reinterpret_cast<bf16_t*>(smem_raw);                           // 8 KB
  bf16_t* h1sw = reinterpret_cast<bf16_t*>(smem_raw + 2 * (L_ * D_));           // 64 KB
  bf16_t* h2sw = reinterpret_cast<bf16_t*>(smem_raw + 2 * (L_ * D_ + L_ * H_)); // 64 KB

  const int g    = blockIdx.x;
  const int n    = blockIdx.y;
  const int tid  = threadIdx.x;
  const int wv   = tid >> 5;
  const int lane = tid & 31;

  const float b1 = beta1[g];
  const float b2 = beta2[g];

  const float* zg  = z  + ((size_t)n * (G_ * D_) + (size_t)g * D_) * L_;  // [D][L]
  const AW*    w1g = w1 + (size_t)g * H_ * D_;
  const AW*    w2g = w2 + (size_t)g * H_ * H_;
  const AW*    w3g = w3 + (size_t)g * D_ * H_;
  float*       og  = out + ((size_t)n * (G_ * D_) + (size_t)g * D_) * L_; // [D][L]

  // ---- Stage x into B-swizzled LDS (bf16 pairs along d, coalesced reads) ----
  for (int idx = tid; idx < (D_ / 2) * L_; idx += 256) {
    const int l  = idx & 63;
    const int d  = (idx >> 6) * 2;
    const float v0 = zg[(size_t)d * L_ + l];
    const float v1 = zg[(size_t)(d + 1) * L_ + l];
    const int tile = (d >> 5) * 4 + (l >> 4);
    const int lb   = (l & 15) + (((d >> 4) & 1) << 4);
    *reinterpret_cast<unsigned int*>(xsw + ((size_t)tile * 32 + lb) * 16 + (d & 15)) =
        pack_bf16x2(v0, v1);
  }
  __syncthreads();

  // ---- GEMM1: h1 = swish_b1(w1 @ x), K = 64 ----
  gemm512_swish(w1g, D_ / 32, xsw, h1sw, b1, wv, lane);
  __syncthreads();

  // ---- GEMM2: h2 = swish_b2(w2 @ h1), K = 512 (A streamed from L2) ----
  gemm512_swish(w2g, H_ / 32, h1sw, h2sw, b2, wv, lane);
  __syncthreads();

  // ---- GEMM3: o = tanh(w3 @ h2), K = 512. Wave: 1 m-tile x 2 n-tiles ----
  {
    const int mt  = wv >> 1;         // 0..3
    const int ntb = (wv & 1) * 2;    // 0 or 2
    const int m0  = mt * 16;
    v8f acc[2] = {};
    for (int ks = 0; ks < H_ / 32; ++ks) {
      v16bf a = load_a_any(w3g, H_ / 32, m0, ks, lane);  // shared by both n-tiles
#pragma unroll
      for (int i = 0; i < 2; ++i) {
        v16bf b = load_frag_sw(h2sw, ks * 4 + ntb + i, lane);
        wmma_bf16(a, b, acc[i]);
      }
    }
    const int mbase = m0 + (lane >> 4) * 8;
#pragma unroll
    for (int i = 0; i < 2; ++i) {
      const int ncol = (ntb + i) * 16 + (lane & 15);
#pragma unroll
      for (int r = 0; r < 8; ++r) {
        og[(size_t)(mbase + r) * L_ + ncol] = tanh_fast(acc[i][r]);
      }
    }
  }
}

extern "C" void kernel_launch(void* const* d_in, const int* in_sizes, int n_in,
                              void* d_out, int out_size, void* d_ws, size_t ws_size,
                              hipStream_t stream) {
  (void)n_in; (void)out_size;
  const float* z  = (const float*)d_in[0];
  const float* w1 = (const float*)d_in[1];
  const float* w2 = (const float*)d_in[2];
  const float* w3 = (const float*)d_in[3];
  const float* b1 = (const float*)d_in[4];
  const float* b2 = (const float*)d_in[5];
  float* out = (float*)d_out;

  const int n_batch = in_sizes[0] / (G_ * D_ * L_);  // 16
  constexpr size_t SMEM = (size_t)2 * (L_ * D_ + L_ * H_ + L_ * H_);  // 136 KB

  constexpr size_t W1E = (size_t)G_ * H_ * D_;
  constexpr size_t W2E = (size_t)G_ * H_ * H_;
  constexpr size_t W3E = (size_t)G_ * D_ * H_;
  constexpr size_t WS_NEEDED = (W1E + W2E + W3E) * sizeof(bf16_t);  // ~80 MB

  dim3 grid(G_, n_batch);

  if (ws_size >= WS_NEEDED) {
    bf16_t* w1s = (bf16_t*)d_ws;
    bf16_t* w2s = w1s + W1E;
    bf16_t* w3s = w2s + W2E;

    convert_swizzle_kernel<<<dim3((unsigned)(H_ * D_ / 2 / 256), G_), 256, 0, stream>>>(
        w1, w1s, D_, 1);  // ktiles=2  -> shift 1
    convert_swizzle_kernel<<<dim3((unsigned)(H_ * H_ / 2 / 256), G_), 256, 0, stream>>>(
        w2, w2s, H_, 4);  // ktiles=16 -> shift 4
    convert_swizzle_kernel<<<dim3((unsigned)(D_ * H_ / 2 / 256), G_), 256, 0, stream>>>(
        w3, w3s, H_, 4);

    (void)hipFuncSetAttribute(
        reinterpret_cast<const void*>(&ode_fused_kernel_t<bf16_t>),
        hipFuncAttributeMaxDynamicSharedMemorySize, (int)SMEM);
    ode_fused_kernel_t<bf16_t><<<grid, 256, SMEM, stream>>>(z, w1s, w2s, w3s, b1, b2,
                                                            out);
  } else {
    (void)hipFuncSetAttribute(
        reinterpret_cast<const void*>(&ode_fused_kernel_t<float>),
        hipFuncAttributeMaxDynamicSharedMemorySize, (int)SMEM);
    ode_fused_kernel_t<float><<<grid, 256, SMEM, stream>>>(z, w1, w2, w3, b1, b2, out);
  }
}
